// Attention_70171175682729
// MI455X (gfx1250) — matrix operand order repaired and checked by reading, not compiled, not run on hardware
//
#include <hip/hip_runtime.h>
#include <hip/hip_bf16.h>

typedef __attribute__((ext_vector_type(16))) _Float16 v16h;
typedef __attribute__((ext_vector_type(8)))  float    v8f;

#define N_SP   1024
#define C_DIM  512
#define HEADS  8
#define DHEAD  64
#define N_GRP  32
#define GN_EPS 1e-6f

// ---------------------------------------------------------------------------
// WMMA fragment loaders (this code's contiguous k = 16*(lane/16) + elem order, wave32)
// A (16x32, fp16): lane = m + 16*(k/16); elem = k%16
//   -> per lane: 16 contiguous halfs at row (lane&15), cols 16*khi
// B (32x16, fp16) stored transposed [n][k] (k contiguous):
//   lane = n + 16*(k/16); elem = k%16 -> 16 contiguous halfs at row (lane&15)
// ---------------------------------------------------------------------------
__device__ inline v16h frag_a32(const _Float16* __restrict__ p, int ld, int lane) {
  const _Float16* row = p + (lane & 15) * ld + ((lane >> 4) & 1) * 16;
  v16h r;
#pragma unroll
  for (int e = 0; e < 16; ++e) {
    int k = e;
    r[e] = row[k];
  }
  return r;
}

__device__ inline v16h frag_bT(const _Float16* __restrict__ p, int ld, int lane) {
  const _Float16* row = p + (lane & 15) * ld + ((lane >> 4) & 1) * 16;
  v16h r;
#pragma unroll
  for (int e = 0; e < 16; ++e) r[e] = row[e];
  return r;
}

// ---------------------------------------------------------------------------
// Kernel 1: GroupNorm over [c/32 x 1024] per (batch, group); fp16 output
// ---------------------------------------------------------------------------
__global__ __launch_bounds__(256)
void gn_kernel(const float* __restrict__ x,
               const float* __restrict__ gamma,
               const float* __restrict__ beta,
               _Float16* __restrict__ h16) {
  const int b = blockIdx.x / N_GRP;
  const int g = blockIdx.x % N_GRP;
  const int CPG = C_DIM / N_GRP;            // 16
  const int TOT = CPG * N_SP;               // 16384
  const float* xb = x + (size_t)b * C_DIM * N_SP + (size_t)g * CPG * N_SP;
  _Float16* hb = h16 + (size_t)b * C_DIM * N_SP + (size_t)g * CPG * N_SP;
  const int t = threadIdx.x;

  float s = 0.f, ss = 0.f;
  for (int i = t; i < TOT; i += 256) {
    float v = xb[i];
    s += v; ss += v * v;
  }
  __shared__ float red[256], red2[256];
  red[t] = s; red2[t] = ss;
  __syncthreads();
  for (int off = 128; off > 0; off >>= 1) {
    if (t < off) { red[t] += red[t + off]; red2[t] += red2[t + off]; }
    __syncthreads();
  }
  __shared__ float s_mean, s_rinv;
  if (t == 0) {
    float mean = red[0] / (float)TOT;
    float var  = red2[0] / (float)TOT - mean * mean;
    s_mean = mean;
    s_rinv = rsqrtf(var + GN_EPS);
  }
  __syncthreads();
  const float mean = s_mean, rinv = s_rinv;
  for (int i = t; i < TOT; i += 256) {
    int ch = g * CPG + (i >> 10);
    float v = (xb[i] - mean) * rinv * gamma[ch] + beta[ch];
    hb[i] = (_Float16)v;
  }
}

// ---------------------------------------------------------------------------
// Kernels 2 & 4: tiled GEMM  out[b][m][n] = sum_k W[m][k] * B[b][k][n] + bias
// 64x64 block tile, 4 waves, wave w owns a 16x64 strip; K-step 32.
// PROJ=false: fp16 output (qkv). PROJ=true: fp32 output + residual.
// ---------------------------------------------------------------------------
template <bool PROJ>
__global__ __launch_bounds__(128)
void gemm_kernel(const float* __restrict__ W,
                 const _Float16* __restrict__ Bm,
                 const float* __restrict__ bias,
                 const float* __restrict__ resid,
                 _Float16* __restrict__ out16,
                 float* __restrict__ out32,
                 int M, int K) {
  const int nb  = blockIdx.x * 64;
  const int mb  = blockIdx.y * 64;
  const int bat = blockIdx.z;
  const _Float16* Bp = Bm + (size_t)bat * K * N_SP;

  __shared__ _Float16 As[64 * 40];   // [m][k] row-major, ld=40 (16B-aligned rows)
  __shared__ _Float16 Bs[64 * 40];   // [n][k] transposed, ld=40

  const int t = threadIdx.x;
  const int w = t >> 5;
  const int lane = t & 31;

  v8f acc[4] = {};

  for (int k0 = 0; k0 < K; k0 += 32) {
    __syncthreads();
#pragma unroll
    for (int i = 0; i < 16; ++i) {           // A: 64x32, fp32 -> fp16
      int idx = t + i * 128;
      int m = idx >> 5, kk = idx & 31;
      As[m * 40 + kk] = (_Float16)W[(size_t)(mb + m) * K + k0 + kk];
    }
#pragma unroll
    for (int i = 0; i < 16; ++i) {           // B: 32x64 -> Bs[n][k]
      int idx = t + i * 128;
      int n = idx & 63, kk = idx >> 6;
      Bs[n * 40 + kk] = Bp[(size_t)(k0 + kk) * N_SP + nb + n];
    }
    __syncthreads();

    v16h a = frag_a32(&As[(w * 16) * 40], 40, lane);
#pragma unroll
    for (int nt = 0; nt < 4; ++nt) {
      v16h bf = frag_bT(&Bs[(nt * 16) * 40], 40, lane);
      acc[nt] = __builtin_amdgcn_wmma_f32_16x16x32_f16(false, a, false, bf,
                                                       (short)0, acc[nt],
                                                       false, false);
    }
  }

  const int mloc = w * 16 + ((lane >> 4) << 3);
#pragma unroll
  for (int nt = 0; nt < 4; ++nt) {
    int n = nb + nt * 16 + (lane & 15);
#pragma unroll
    for (int e = 0; e < 8; ++e) {
      int m = mb + mloc + e;
      float vv = acc[nt][e] + bias[m];
      size_t oidx = ((size_t)bat * M + m) * N_SP + n;
      if (PROJ) out32[oidx] = vv + resid[oidx];
      else      out16[oidx] = (_Float16)vv;
    }
  }
}

// ---------------------------------------------------------------------------
// Kernel 3: flash-style attention per (batch, head, 64-query tile)
// 4 waves; wave w owns 16 query rows. Online softmax in fp32.
// ---------------------------------------------------------------------------
__global__ __launch_bounds__(128)
void attn_kernel(const _Float16* __restrict__ qkv, _Float16* __restrict__ att) {
  const int it   = blockIdx.x;     // query tile 0..15
  const int head = blockIdx.y;
  const int bat  = blockIdx.z;
  const int ib   = it * 64;
  const size_t base = (size_t)bat * 3 * C_DIM * N_SP + (size_t)head * DHEAD * N_SP;
  const _Float16* q = qkv + base;
  const _Float16* k = qkv + base + (size_t)C_DIM * N_SP;
  const _Float16* v = qkv + base + (size_t)2 * C_DIM * N_SP;

  __shared__ _Float16 Qs[64 * 72];     // [i][d]   (Q transposed)
  __shared__ _Float16 Ks[64 * 72];     // [j][d]   (K transposed: B-layout for S)
  __shared__ _Float16 Vs[64 * 72];     // [d][j]   (B-layout for O)
  __shared__ _Float16 Ps[4][16 * 72];  // per-wave P tile [i][j]

  const int t = threadIdx.x, w = t >> 5, lane = t & 31;
  const float scale = 0.125f;          // 1/sqrt(64)

#pragma unroll
  for (int xx = 0; xx < 32; ++xx) {    // load Q tile transposed (coalesced reads)
    int idx = t + xx * 128;
    int i = idx & 63, d = idx >> 6;
    Qs[i * 72 + d] = q[(size_t)d * N_SP + ib + i];
  }

  float rmax[8], rsum[8];
#pragma unroll
  for (int e = 0; e < 8; ++e) { rmax[e] = -1e30f; rsum[e] = 0.f; }
  v8f accO[4] = {};

  for (int jt = 0; jt < 16; ++jt) {
    const int jb = jt * 64;
    __syncthreads();
#pragma unroll
    for (int xx = 0; xx < 32; ++xx) {
      int idx = t + xx * 128;
      int j = idx & 63, d = idx >> 6;
      Ks[j * 72 + d] = k[(size_t)d * N_SP + jb + j];
      Vs[d * 72 + j] = v[(size_t)d * N_SP + jb + j];
    }
    __syncthreads();

    // S = Q_i * K_j^T   (16x64 per wave)
    v8f accS[4] = {};
#pragma unroll
    for (int kk = 0; kk < 64; kk += 32) {
      v16h aq = frag_a32(&Qs[(w * 16) * 72 + kk], 72, lane);
#pragma unroll
      for (int nt = 0; nt < 4; ++nt) {
        v16h bk = frag_bT(&Ks[(nt * 16) * 72 + kk], 72, lane);
        accS[nt] = __builtin_amdgcn_wmma_f32_16x16x32_f16(false, aq, false, bk,
                                                          (short)0, accS[nt],
                                                          false, false);
      }
    }

    // online softmax: row m = 8*(lane/16)+e, replicated over 16 lanes
    float mloc[8];
#pragma unroll
    for (int e = 0; e < 8; ++e) {
      float mv = -1e30f;
#pragma unroll
      for (int nt = 0; nt < 4; ++nt) mv = fmaxf(mv, accS[nt][e] * scale);
      mloc[e] = mv;
    }
#pragma unroll
    for (int msk = 1; msk < 16; msk <<= 1)
#pragma unroll
      for (int e = 0; e < 8; ++e)
        mloc[e] = fmaxf(mloc[e], __shfl_xor(mloc[e], msk, 32));

    float alpha[8], psum[8];
#pragma unroll
    for (int e = 0; e < 8; ++e) {
      float mn = fmaxf(rmax[e], mloc[e]);
      alpha[e] = __expf(rmax[e] - mn);
      rmax[e] = mn;
      psum[e] = 0.f;
    }
#pragma unroll
    for (int nt = 0; nt < 4; ++nt) {
#pragma unroll
      for (int e = 0; e < 8; ++e) {
        float p = __expf(accS[nt][e] * scale - rmax[e]);
        psum[e] += p;
        int row = ((lane >> 4) << 3) + e;
        Ps[w][row * 72 + nt * 16 + (lane & 15)] = (_Float16)p;  // D->A relayout
      }
    }
#pragma unroll
    for (int msk = 1; msk < 16; msk <<= 1)
#pragma unroll
      for (int e = 0; e < 8; ++e)
        psum[e] += __shfl_xor(psum[e], msk, 32);
#pragma unroll
    for (int e = 0; e < 8; ++e) rsum[e] = rsum[e] * alpha[e] + psum[e];
#pragma unroll
    for (int nt = 0; nt < 4; ++nt)
#pragma unroll
      for (int e = 0; e < 8; ++e) accO[nt][e] *= alpha[e];

    // O += P * V^T
#pragma unroll
    for (int kk = 0; kk < 64; kk += 32) {
      v16h ap = frag_a32(&Ps[w][kk], 72, lane);
#pragma unroll
      for (int dt = 0; dt < 4; ++dt) {
        v16h bv = frag_bT(&Vs[(dt * 16) * 72 + kk], 72, lane);
        accO[dt] = __builtin_amdgcn_wmma_f32_16x16x32_f16(false, ap, false, bv,
                                                          (short)0, accO[dt],
                                                          false, false);
      }
    }
  }

  // normalize + store: att[b][head*64 + d][i]  (layout "(h d) n")
#pragma unroll
  for (int dt = 0; dt < 4; ++dt) {
    int d = dt * 16 + (lane & 15);
#pragma unroll
    for (int e = 0; e < 8; ++e) {
      int irow = ib + w * 16 + ((lane >> 4) << 3) + e;
      float vv = accO[dt][e] / rsum[e];
      att[((size_t)bat * C_DIM + head * DHEAD + d) * N_SP + irow] = (_Float16)vv;
    }
  }
}

// ---------------------------------------------------------------------------
extern "C" void kernel_launch(void* const* d_in, const int* in_sizes, int n_in,
                              void* d_out, int out_size, void* d_ws, size_t ws_size,
                              hipStream_t stream) {
  (void)in_sizes; (void)n_in; (void)out_size; (void)ws_size;
  const float* x        = (const float*)d_in[0];
  const float* gn_gamma = (const float*)d_in[1];
  const float* gn_beta  = (const float*)d_in[2];
  const float* w_qkv    = (const float*)d_in[3];
  const float* b_qkv    = (const float*)d_in[4];
  const float* w_out    = (const float*)d_in[5];
  const float* b_out    = (const float*)d_in[6];
  float* out = (float*)d_out;

  // workspace: h16 (8 MB) | qkv16 (24 MB); attention output reuses h16 region
  _Float16* h16   = (_Float16*)d_ws;
  _Float16* qkv16 = h16 + (size_t)8 * C_DIM * N_SP;
  _Float16* att16 = h16;

  gn_kernel<<<8 * N_GRP, 256, 0, stream>>>(x, gn_gamma, gn_beta, h16);

  gemm_kernel<false><<<dim3(N_SP / 64, (3 * C_DIM) / 64, 8), 128, 0, stream>>>(
      w_qkv, h16, b_qkv, nullptr, qkv16, nullptr, 3 * C_DIM, C_DIM);

  attn_kernel<<<dim3(N_SP / 64, HEADS, 8), 128, 0, stream>>>(qkv16, att16);

  gemm_kernel<true><<<dim3(N_SP / 64, C_DIM / 64, 8), 128, 0, stream>>>(
      w_out, att16, b_out, x, nullptr, out, C_DIM, C_DIM);
}
